// VisionTritonAttention_49246095016314
// MI455X (gfx1250) — compile-verified, matrix operand order checked by reading, not run on hardware
//
#include <hip/hip_runtime.h>
#include <math.h>

typedef __attribute__((ext_vector_type(16))) _Float16 v16h;
typedef __attribute__((ext_vector_type(8)))  _Float16 v8h;
typedef __attribute__((ext_vector_type(8)))  float    v8f;

#define B_    8
#define S_    1024
#define H_    16
#define D_    64
#define QTILE 128     // 8 waves x 16 q-rows per workgroup
#define KSTEP 32      // keys per pipeline step (== WMMA K for f16)

#define KSTRIDE 72    // halves per K-tile row (64 + 8 pad) -> conflict-free b128 reads
#define VSTRIDE 40    // halves per Vt row   (32 + 8 pad)
#define PSTRIDE 40    // halves per P-stage row

// ---- DPP16 xor-butterfly reductions within each 16-lane half-wave ----
// xor1 = quad_perm[1,0,3,2] (0xB1), xor2 = quad_perm[2,3,0,1] (0x4E),
// xor4 = row_half_mirror (0x141),  xor8 = row_mirror (0x140)
template <int CTRL>
__device__ __forceinline__ float dpp_xchg(float x) {
  return __int_as_float(
      __builtin_amdgcn_mov_dpp(__float_as_int(x), CTRL, 0xF, 0xF, true));
}
__device__ __forceinline__ float rowmax16(float x) {
  x = fmaxf(x, dpp_xchg<0xB1>(x));
  x = fmaxf(x, dpp_xchg<0x4E>(x));
  x = fmaxf(x, dpp_xchg<0x141>(x));
  x = fmaxf(x, dpp_xchg<0x140>(x));
  return x;
}
__device__ __forceinline__ float rowsum16(float x) {
  x += dpp_xchg<0xB1>(x);
  x += dpp_xchg<0x4E>(x);
  x += dpp_xchg<0x141>(x);
  x += dpp_xchg<0x140>(x);
  return x;
}

// Raw v_exp_f32 (args are always <= 0 here; underflow flushing to 0 is fine)
__device__ __forceinline__ float ex2(float x) { return __builtin_amdgcn_exp2f(x); }

__device__ __forceinline__ v8f wmma_f16(v16h a, v16h b, v8f c) {
  // D(f32 16x16) = A(16x32 f16) * B(32x16 f16) + C
  return __builtin_amdgcn_wmma_f32_16x16x32_f16(
      /*neg_a=*/false, a, /*neg_b=*/false, b,
      /*c_mod=*/(short)0, c, /*reuse_a=*/false, /*reuse_b=*/false);
}

__global__ __launch_bounds__(256)
void VisionTritonAttention_49246095016314_kernel(
    const float* __restrict__ q, const float* __restrict__ k,
    const float* __restrict__ v, const int* __restrict__ cu,
    float* __restrict__ out) {
  // double-buffered K/V tiles + per-wave P staging (~29 KB total)
  __shared__ __align__(16) _Float16 Klds[2][KSTEP * KSTRIDE];
  __shared__ __align__(16) _Float16 Vt[2][D_ * VSTRIDE];
  __shared__ __align__(16) _Float16 Pst[8 * 16 * PSTRIDE];

  const int tid  = threadIdx.x;
  const int wave = tid >> 5;
  const int lane = tid & 31;
  const int lo   = lane & 15;   // column / row-in-half selector
  const int hi   = lane >> 4;   // which 16-lane half of the wave

  const int nqb = S_ / QTILE;
  const int wg  = blockIdx.x;
  const int qb  = wg % nqb;
  const int h   = (wg / nqb) % H_;
  const int b   = wg / (nqb * H_);

  const int len = cu[b + 1] - cu[b];
  // softmax in exp2 domain: fold (1/sqrt(D)) * log2(e) into Q
  const float scale = 0.125f * 1.44269504088896340736f;

  const size_t tokStride = (size_t)H_ * D_;

  // ---------- Q A-fragments (16x32 f16, two K=32 chunks over D=64) ----------
  // A layout: lane L holds row L%16; halves j<8 -> K = j + 8*hi; j>=8 -> K = j + 8 + 8*hi
  const int qrow = qb * QTILE + wave * 16 + lo;
  const float* qp = q + ((size_t)(b * S_ + qrow) * H_ + h) * D_;
  v16h aQ[2];
  #pragma unroll
  for (int dc = 0; dc < 2; ++dc) {
    float qtmp[16];
    const float* p0 = qp + dc * 32 + 8 * hi;        // K run j=0..7
    const float* p1 = qp + dc * 32 + 16 + 8 * hi;   // K run j=8..15
    *(float4*)&qtmp[0]  = *(const float4*)(p0);
    *(float4*)&qtmp[4]  = *(const float4*)(p0 + 4);
    *(float4*)&qtmp[8]  = *(const float4*)(p1);
    *(float4*)&qtmp[12] = *(const float4*)(p1 + 4);
    #pragma unroll
    for (int j = 0; j < 16; ++j) aQ[dc][j] = (_Float16)(qtmp[j] * scale);
  }

  // ---------- accumulators ----------
  v8f O[4];                         // O tiles over D columns t*16..t*16+15
  #pragma unroll
  for (int t = 0; t < 4; ++t)
    #pragma unroll
    for (int r = 0; r < 8; ++r) O[t][r] = 0.0f;
  float m_i[8], l_i[8];             // per-row (r + 8*hi): running max, LANE-LOCAL partial sum
  #pragma unroll
  for (int r = 0; r < 8; ++r) { m_i[r] = -3.0e38f; l_i[r] = 0.0f; }

  // cooperative-load assignment: 256 thr * 8 f32 = one 32x64 tile
  const int ckey = tid >> 3;            // 0..31
  const int cd   = (tid & 7) * 8;       // 0,8,...,56
  const float* kbase = k + ((size_t)(b * S_) * H_ + h) * D_ + (size_t)ckey * tokStride + cd;
  const float* vbase = v + ((size_t)(b * S_) * H_ + h) * D_ + (size_t)ckey * tokStride + cd;

  _Float16* prow_base = &Pst[wave * 16 * PSTRIDE];

  // only process key blocks that contain valid keys (uniform across WG)
  const int kend   = min(S_, (len + KSTEP - 1) & ~(KSTEP - 1));
  const int nsteps = kend / KSTEP;

  // ---------- prologue: stage tile 0 into buffer 0 ----------
  if (nsteps > 0) {
    float ktmp[8], vtmp[8];
    *(float4*)&ktmp[0] = *(const float4*)(kbase);
    *(float4*)&ktmp[4] = *(const float4*)(kbase + 4);
    *(float4*)&vtmp[0] = *(const float4*)(vbase);
    *(float4*)&vtmp[4] = *(const float4*)(vbase + 4);
    v8h hk;
    #pragma unroll
    for (int j = 0; j < 8; ++j) hk[j] = (_Float16)ktmp[j];
    *(v8h*)&Klds[0][ckey * KSTRIDE + cd] = hk;
    #pragma unroll
    for (int j = 0; j < 8; ++j)
      Vt[0][(cd + j) * VSTRIDE + ckey] = (_Float16)vtmp[j];
  }

  #pragma unroll 1
  for (int i = 0; i < nsteps; ++i) {
    const int kb  = i * KSTEP;
    const int cur = i & 1;
    const bool have_next = (i + 1) < nsteps;

    __syncthreads();   // buf[cur] fully staged; buf[1-cur] readers (iter i-1) done

    // ---- issue next tile's global loads early (overlap with compute) ----
    float kn[8], vn[8];
    if (have_next) {
      const float* kr = kbase + (size_t)(kb + KSTEP) * tokStride;
      const float* vr = vbase + (size_t)(kb + KSTEP) * tokStride;
      *(float4*)&kn[0] = *(const float4*)(kr);
      *(float4*)&kn[4] = *(const float4*)(kr + 4);
      *(float4*)&vn[0] = *(const float4*)(vr);
      *(float4*)&vn[4] = *(const float4*)(vr + 4);
      if (kb + 2 * KSTEP < kend) {              // global_prefetch_b8 two tiles ahead
        __builtin_prefetch(kr + (size_t)KSTEP * tokStride, 0, 0);
        __builtin_prefetch(vr + (size_t)KSTEP * tokStride, 0, 0);
      }
    }

    // ---- K B-fragments: B[kd][n], column n = key (t*16 + lo) ----
    v16h bK[2][2];
    #pragma unroll
    for (int t = 0; t < 2; ++t) {
      const _Float16* krow0 = &Klds[cur][(t * 16 + lo) * KSTRIDE];
      #pragma unroll
      for (int dc = 0; dc < 2; ++dc) {
        v8h x0 = *(const v8h*)(krow0 + dc * 32 + 8 * hi);
        v8h x1 = *(const v8h*)(krow0 + dc * 32 + 16 + 8 * hi);
        #pragma unroll
        for (int j = 0; j < 8; ++j) { bK[dc][t][j] = x0[j]; bK[dc][t][8 + j] = x1[j]; }
      }
    }

    // ---- scores: two 16x16 tiles over this 32-key block ----
    v8f s0 = {}; v8f s1 = {};
    s0 = wmma_f16(aQ[0], bK[0][0], s0);
    s0 = wmma_f16(aQ[1], bK[1][0], s0);
    s1 = wmma_f16(aQ[0], bK[0][1], s1);
    s1 = wmma_f16(aQ[1], bK[1][1], s1);

    // ---- online softmax (exp2 domain); stage masked P (f16) to LDS ----
    // l_i is kept lane-local (per-column partial); reduced once in epilogue.
    if (kb + KSTEP <= len) {
      // fast path: every key in this block is valid
      #pragma unroll
      for (int r = 0; r < 8; ++r) {  // row = r + 8*hi
        float mx = rowmax16(fmaxf(s0[r], s1[r]));
        float mn = fmaxf(m_i[r], mx);
        float corr = ex2(m_i[r] - mn);
        m_i[r] = mn;
        float p0 = ex2(s0[r] - mn);
        float p1 = ex2(s1[r] - mn);
        l_i[r] = l_i[r] * corr + (p0 + p1);
        #pragma unroll
        for (int t = 0; t < 4; ++t) O[t][r] *= corr;
        _Float16* pr = prow_base + (r + 8 * hi) * PSTRIDE;
        pr[lo]      = (_Float16)p0;
        pr[16 + lo] = (_Float16)p1;
      }
    } else {
      // tail path: mask out-of-range keys
      const int n0 = kb + lo;
      const int n1 = kb + 16 + lo;
      #pragma unroll
      for (int r = 0; r < 8; ++r) {
        float mx = rowmax16(fmaxf(s0[r], s1[r]));
        float mn = fmaxf(m_i[r], mx);
        float corr = ex2(m_i[r] - mn);
        m_i[r] = mn;
        float p0 = (n0 < len) ? ex2(s0[r] - mn) : 0.0f;
        float p1 = (n1 < len) ? ex2(s1[r] - mn) : 0.0f;
        l_i[r] = l_i[r] * corr + (p0 + p1);
        #pragma unroll
        for (int t = 0; t < 4; ++t) O[t][r] *= corr;
        _Float16* pr = prow_base + (r + 8 * hi) * PSTRIDE;
        pr[lo]      = (_Float16)p0;
        pr[16 + lo] = (_Float16)p1;
      }
    }

    // ---- P A-fragment (16x32) from per-wave staging ----
    v16h aP;
    {
      const _Float16* pm = prow_base + lo * PSTRIDE;
      v8h pl = *(const v8h*)(pm + 8 * hi);
      v8h ph = *(const v8h*)(pm + 16 + 8 * hi);
      #pragma unroll
      for (int j = 0; j < 8; ++j) { aP[j] = pl[j]; aP[8 + j] = ph[j]; }
    }

    // ---- O += P * V : 4 WMMA over D columns ----
    #pragma unroll
    for (int t = 0; t < 4; ++t) {
      const _Float16* vrow = &Vt[cur][(t * 16 + lo) * VSTRIDE];
      v8h y0 = *(const v8h*)(vrow + 8 * hi);
      v8h y1 = *(const v8h*)(vrow + 16 + 8 * hi);
      v16h bV;
      #pragma unroll
      for (int j = 0; j < 8; ++j) { bV[j] = y0[j]; bV[8 + j] = y1[j]; }
      O[t] = wmma_f16(aP, bV, O[t]);
    }

    // ---- stage next tile into the other buffer (safe after top barrier) ----
    if (have_next) {
      const int nxt = 1 - cur;
      v8h hk;
      #pragma unroll
      for (int j = 0; j < 8; ++j) hk[j] = (_Float16)kn[j];
      *(v8h*)&Klds[nxt][ckey * KSTRIDE + cd] = hk;
      #pragma unroll
      for (int j = 0; j < 8; ++j)
        Vt[nxt][(cd + j) * VSTRIDE + ckey] = (_Float16)vn[j];
    }
  }

  // ---------- epilogue: reduce l, normalize, store ----------
  float inv[8];
  #pragma unroll
  for (int r = 0; r < 8; ++r) {
    float lsum = rowsum16(l_i[r]);   // deferred cross-lane reduction
    inv[r] = (lsum > 0.0f) ? (1.0f / lsum) : 0.0f;
  }

  const int orow_base = qb * QTILE + wave * 16 + 8 * hi;
  #pragma unroll
  for (int r = 0; r < 8; ++r) {
    const int orow = orow_base + r;
    float* op = out + ((size_t)(b * S_ + orow) * H_ + h) * D_;
    const bool rvalid = orow < len;
    #pragma unroll
    for (int t = 0; t < 4; ++t) {
      float val = rvalid ? O[t][r] * inv[r] : 0.0f;
      op[t * 16 + lo] = val;
    }
  }
}

extern "C" void kernel_launch(void* const* d_in, const int* in_sizes, int n_in,
                              void* d_out, int out_size, void* d_ws, size_t ws_size,
                              hipStream_t stream) {
  (void)in_sizes; (void)n_in; (void)d_ws; (void)ws_size; (void)out_size;
  const float* q  = (const float*)d_in[0];
  const float* k  = (const float*)d_in[1];
  const float* v  = (const float*)d_in[2];
  const int*   cu = (const int*)d_in[3];
  // d_in[4] = max_seqlen (unused; S_ fixed by problem shape)
  float* out = (float*)d_out;

  dim3 grid(B_ * H_ * (S_ / QTILE));   // 1024 workgroups
  dim3 block(256);                     // 8 wave32
  VisionTritonAttention_49246095016314_kernel<<<grid, block, 0, stream>>>(q, k, v, cu, out);
}
